// ScaledDotProductAttention_5334349382388
// MI455X (gfx1250) — matrix operand order repaired and checked by reading, not compiled, not run on hardware
//
#include <hip/hip_runtime.h>
#include <hip/hip_bf16.h>
#include <stdint.h>

typedef __attribute__((ext_vector_type(16))) __bf16 v16bf;
typedef __attribute__((ext_vector_type(8)))  float  v8f;

#define N_SEQ  2048
#define D_HEAD 64
#define NEGBIG (-1.0e30f)

// xor-shuffle within 16-lane groups via ds_swizzle (group-of-32 mode:
// imm = xor_mask<<10 | or_mask<<5 | and_mask; and=0x1f, or=0)
#define SWZ_XOR(x, m) \
  __int_as_float(__builtin_amdgcn_ds_swizzle(__float_as_int(x), (((m) << 10) | 0x1f)))

__device__ __forceinline__ float grp16_max(float x) {
  x = fmaxf(x, SWZ_XOR(x, 1));
  x = fmaxf(x, SWZ_XOR(x, 2));
  x = fmaxf(x, SWZ_XOR(x, 4));
  x = fmaxf(x, SWZ_XOR(x, 8));
  return x;
}
__device__ __forceinline__ float grp16_sum(float x) {
  x += SWZ_XOR(x, 1);
  x += SWZ_XOR(x, 2);
  x += SWZ_XOR(x, 4);
  x += SWZ_XOR(x, 8);
  return x;
}

// K-dim base inside a 16x32 bf16 A-fragment for VGPR-pair vr and lane half hi
// (ISA 7.12.2: lanes0-15 V0..3 -> K 0..7, V4..7 -> K16..23; lanes16-31 +8)
__device__ __forceinline__ int a_kbase(int vr, int hi) {
  return ((vr < 4) ? (2 * vr) : (8 + 2 * vr)) + (hi ? 8 : 0);
}

// Async global->LDS copy of 16B per lane (CDNA5 TDM-class path, ASYNCcnt).
__device__ __forceinline__ void async_b128(const void* gsrc, const void* ldst) {
  uint32_t lds_off = (uint32_t)(uintptr_t)ldst;   // generic LDS ptr low 32b = LDS offset
  uint64_t gaddr   = (uint64_t)(uintptr_t)gsrc;
  asm volatile("global_load_async_to_lds_b128 %0, %1, off"
               :: "v"(lds_off), "v"(gaddr) : "memory");
}

__global__ __launch_bounds__(128)
void fa_wmma_kernel(const float* __restrict__ q,
                    const float* __restrict__ kmat,
                    const float* __restrict__ vmat,
                    const float* __restrict__ edge,
                    const unsigned char* __restrict__ mask,
                    float* __restrict__ out) {
  // double-buffered async staging tiles (f32 K/V, u8 mask) + per-wave P tile
  __shared__ float         sK32[2 * 32 * 64];     // [buf][kvrow][d]   16 KB
  __shared__ float         sV32[2 * 32 * 64];     // [buf][kvrow][d]   16 KB
  __shared__ unsigned char sMsk[2 * 64 * 32];     // [buf][qrow][kvloc] 4 KB
  __shared__ __bf16        sP[4 * 16 * 32];       // [wave][m][kvloc]   4 KB

  const int tid  = threadIdx.x;
  const int lane = tid & 31;
  const int wave = tid >> 5;
  const int hi   = lane >> 4;   // lane half (0: lanes0-15, 1: lanes16-31)
  const int ln   = lane & 15;

  const int b     = blockIdx.x >> 5;       // batch (16)
  const int rblk  = blockIdx.x & 31;       // 64-row block within batch
  const int m0blk = rblk * 64;             // block's first row
  const int m0w   = m0blk + wave * 16;     // this wave's first row
  const bool use_mask = (b >= 2);

  const float* kbat = kmat + (size_t)b * N_SEQ * D_HEAD;
  const float* vbat = vmat + (size_t)b * N_SEQ * D_HEAD;
  const unsigned char* mbat = mask + (size_t)b * N_SEQ * N_SEQ;

  // ---- Q fragments: 16x64 rows -> two 16x32 bf16 A-frags, fold in 1/SCALE
  v16bf aq[2];
  {
    const float* qrow = q + ((size_t)b * N_SEQ + (size_t)(m0w + ln)) * D_HEAD;
#pragma unroll
    for (int kb = 0; kb < 2; ++kb) {
#pragma unroll
      for (int vr = 0; vr < 8; ++vr) {
        int kk = a_kbase(vr, hi) + kb * 32;
        aq[kb][2 * vr + 0] = (__bf16)(qrow[kk + 0] * 0.125f);
        aq[kb][2 * vr + 1] = (__bf16)(qrow[kk + 1] * 0.125f);
      }
    }
  }

  float m_i[8], l_i[8];
  v8f o[4];
  {
    v8f z = {0.f,0.f,0.f,0.f,0.f,0.f,0.f,0.f};
#pragma unroll
    for (int f = 0; f < 4; ++f) o[f] = z;
#pragma unroll
    for (int r = 0; r < 8; ++r) { m_i[r] = NEGBIG; l_i[r] = 0.f; }
  }

  __bf16* Pw = sP + wave * (16 * 32);

  // ---- async stage of one 32-row KV tile (+mask tile) into buffer `buf`
  auto stage = [&](int buf, int jt) {
    const float* kt = kbat + (size_t)jt * D_HEAD;
    const float* vt = vbat + (size_t)jt * D_HEAD;
    float* dK = sK32 + buf * (32 * 64);
    float* dV = sV32 + buf * (32 * 64);
#pragma unroll
    for (int i = 0; i < 4; ++i) {
      int idx = i * 128 + tid;               // 512 x 16B chunks per tile
      async_b128(kt + idx * 4, dK + idx * 4);
      async_b128(vt + idx * 4, dV + idx * 4);
    }
    if (use_mask) {                          // 64 rows x 32 cols of u8 = 2 KB
      int row = tid >> 1, half = tid & 1;
      async_b128(mbat + (size_t)(m0blk + row) * N_SEQ + jt + half * 16,
                 sMsk + buf * (64 * 32) + row * 32 + half * 16);
    }
  };

  stage(0, 0);   // prologue: tile 0 in flight

  for (int jt = 0; jt < N_SEQ; jt += 32) {
    const int cur = (jt >> 5) & 1;
    asm volatile("s_wait_asynccnt 0x0" ::: "memory");  // this wave's DMAs done
    __syncthreads();                                   // all waves' tiles visible
    if (jt + 32 < N_SEQ) stage(1 - cur, jt + 32);      // refill other buffer

    const float*         Kc = sK32 + cur * (32 * 64);
    const float*         Vc = sV32 + cur * (32 * 64);
    const unsigned char* Mc = sMsk + cur * (64 * 32);

    // ---- S = (Q/8) * K^T : two 16-col score tiles, K-dim 64 = 2x32
    v8f s[2];
#pragma unroll
    for (int n0 = 0; n0 < 2; ++n0) {
      v8f c = {0.f,0.f,0.f,0.f,0.f,0.f,0.f,0.f};
#pragma unroll
      for (int kb = 0; kb < 2; ++kb) {
        v16bf bk;   // B-frag: N = ln, K = kb*32 + 8*hi + (e < 8 ? e : e + 8)
        const float* kp = Kc + (n0 * 16 + ln) * 64 + kb * 32 + hi * 8;
#pragma unroll
        for (int e = 0; e < 16; ++e) bk[e] = (__bf16)kp[e < 8 ? e : e + 8];
        c = __builtin_amdgcn_wmma_f32_16x16x32_bf16(false, aq[kb], false, bk,
                                                    (short)0, c, false, false);
      }
      s[n0] = c;
    }

    // ---- mask fill / edge-matrix override for batches 0 and 1
    if (!use_mask) {
#pragma unroll
      for (int n0 = 0; n0 < 2; ++n0)
#pragma unroll
        for (int r = 0; r < 8; ++r) {
          int gm = m0w + r + 8 * hi;
          int gc = jt + n0 * 16 + ln;
          s[n0][r] = edge[(size_t)gm * N_SEQ + gc];
        }
    } else {
#pragma unroll
      for (int n0 = 0; n0 < 2; ++n0)
#pragma unroll
        for (int r = 0; r < 8; ++r) {
          int mrow = wave * 16 + r + 8 * hi;           // row within block
          float sv = s[n0][r];
          s[n0][r] = Mc[mrow * 32 + n0 * 16 + ln] ? NEGBIG : sv;
        }
    }

    // ---- online softmax update; stash P (bf16) to LDS for layout conversion
#pragma unroll
    for (int r = 0; r < 8; ++r) {
      float rm   = grp16_max(fmaxf(s[0][r], s[1][r]));
      float mn   = fmaxf(m_i[r], rm);
      float corr = __expf(m_i[r] - mn);
      float p0   = __expf(s[0][r] - mn);
      float p1   = __expf(s[1][r] - mn);
      l_i[r] = l_i[r] * corr + grp16_sum(p0 + p1);
      m_i[r] = mn;
#pragma unroll
      for (int f = 0; f < 4; ++f) o[f][r] *= corr;
      int mrow = r + 8 * hi;   // C-layout: M = r + 8*(lane/16), N = ln
      Pw[mrow * 32 + ln]      = (__bf16)p0;
      Pw[mrow * 32 + 16 + ln] = (__bf16)p1;
    }
    asm volatile("s_wait_dscnt 0x0" ::: "memory");  // same-wave DS RAW

    // ---- O += P(16x32) * V(32x64): A-frag from LDS, 4 B-frags over D
    v16bf aP;
#pragma unroll
    for (int vr = 0; vr < 8; ++vr) {
      int kk = a_kbase(vr, hi);
      aP[2 * vr + 0] = Pw[ln * 32 + kk + 0];
      aP[2 * vr + 1] = Pw[ln * 32 + kk + 1];
    }
#pragma unroll
    for (int f = 0; f < 4; ++f) {
      v16bf bv;   // N = d = f*16 + ln, K = 8*hi + (e < 8 ? e : e + 8) (kv-local row)
#pragma unroll
      for (int e = 0; e < 16; ++e)
        bv[e] = (__bf16)Vc[(hi * 8 + (e < 8 ? e : e + 8)) * 64 + f * 16 + ln];
      o[f] = __builtin_amdgcn_wmma_f32_16x16x32_bf16(false, aP, false, bv,
                                                     (short)0, o[f], false, false);
    }
  }

  // ---- epilogue: normalize by l_i and store [B,N,D] f32
#pragma unroll
  for (int r = 0; r < 8; ++r) {
    int gm = m0w + r + 8 * hi;
    float inv = 1.0f / l_i[r];
    float* orow = out + ((size_t)b * N_SEQ + gm) * D_HEAD;
#pragma unroll
    for (int f = 0; f < 4; ++f) orow[f * 16 + ln] = o[f][r] * inv;
  }
}

extern "C" void kernel_launch(void* const* d_in, const int* in_sizes, int n_in,
                              void* d_out, int out_size, void* d_ws, size_t ws_size,
                              hipStream_t stream) {
  (void)in_sizes; (void)n_in; (void)out_size; (void)d_ws; (void)ws_size;
  const float*         q    = (const float*)d_in[0];
  const float*         kmat = (const float*)d_in[1];
  const float*         vmat = (const float*)d_in[2];
  const float*         edge = (const float*)d_in[3];
  const unsigned char* mask = (const unsigned char*)d_in[4];  // bool array
  float* out = (float*)d_out;

  dim3 grid(16 * (N_SEQ / 64));  // 16 batches x 32 row-blocks = 512
  dim3 block(128);               // 4 waves of 32
  hipLaunchKernelGGL(fa_wmma_kernel, grid, block, 0, stream,
                     q, kmat, vmat, edge, mask, out);
}
